// BiLstmCrf_15436112462311
// MI455X (gfx1250) — matrix operand order repaired and checked by reading, not compiled, not run on hardware
//
#include <hip/hip_runtime.h>

typedef __attribute__((ext_vector_type(16))) _Float16 v16h;
typedef __attribute__((ext_vector_type(8)))  _Float16 v8h;
typedef __attribute__((ext_vector_type(8)))  float    v8f;
typedef __attribute__((ext_vector_type(4)))  unsigned int u32x4;
typedef __attribute__((ext_vector_type(8)))  int          i32x8;
typedef __attribute__((ext_vector_type(4)))  int          i32x4;

constexpr int kB  = 128;   // batch
constexpr int kS  = 512;   // sequence length
constexpr int kE  = 100;   // embedding dim
constexpr int kH  = 128;   // hidden per direction
constexpr int kG  = 512;   // 4*H gates
constexpr int kH2 = 256;   // bidirectional hidden
constexpr int kT  = 32;    // tags
constexpr int kLPAD = 136; // padded LDS row stride (halves); 272B, 16B aligned
constexpr size_t kSB = (size_t)kS * kB;   // 65536 rows

#define HAVE_TDM __has_builtin(__builtin_amdgcn_tensor_load_to_lds)

// ---------------- WMMA helpers (CDNA5 wave32, 16x16x32 f16 -> f32) ----------

__device__ __forceinline__ v8f wmma_f16(v16h a, v16h b, v8f c) {
  return __builtin_amdgcn_wmma_f32_16x16x32_f16(false, a, false, b,
                                                (short)0, c, false, false);
}

__device__ __forceinline__ v8f zero8() {
  v8f z;
#pragma unroll
  for (int i = 0; i < 8; ++i) z[i] = 0.f;
  return z;
}

// A fragment: 16x32 f16 tile, row-major source with leading dim `ld` (halves).
// ISA layout: lanes 0-15 (M=lane): K=0..7,16..23 ; lanes 16-31 (M=lane-16): K=8..15,24..31
__device__ __forceinline__ v16h load_a_frag(const _Float16* __restrict__ p0, int ld) {
  const int lane = threadIdx.x & 31;
  const int m  = lane & 15;
  const int kb = (lane >> 4) << 3;          // 0 or 8
  const _Float16* p = p0 + (size_t)m * ld + kb;
  v8h lo = *(const v8h*)(p);                // K = kb .. kb+7
  v8h hi = *(const v8h*)(p + 16);           // K = 16+kb .. 16+kb+7
  return __builtin_shufflevector(lo, hi, 0,1,2,3,4,5,6,7,8,9,10,11,12,13,14,15);
}

// B fragment: 32x16 f16 tile, sourced from W^T row-major (row = n, col = k, ld halves).
// ISA layout: lanes 0-15 (N=lane): K=0..7,16..23 ; lanes 16-31 (N=lane-16): K=8..15,24..31
__device__ __forceinline__ v16h load_b_frag(const _Float16* __restrict__ p0, int ld) {
  const int lane = threadIdx.x & 31;
  const int n  = lane & 15;
  const int kb = (lane >> 4) << 3;          // 0 or 8
  const _Float16* p = p0 + (size_t)n * ld + kb;
  v8h lo = *(const v8h*)(p);
  v8h hi = *(const v8h*)(p + 16);
  return __builtin_shufflevector(lo, hi, 0,1,2,3,4,5,6,7,8,9,10,11,12,13,14,15);
}

__device__ __forceinline__ float sigmoidf_(float x) {
  return 1.f / (1.f + __expf(-x));
}

// ------------- Tensor Data Mover: 2-D tile load Global -> LDS ---------------
// Builds a D# per CDNA5 ISA and issues TENSOR_LOAD_TO_LDS.
// dim0_elems: row length (f16 elements), dim1_rows: #rows, stride0_elems: row
// stride in memory. Optional LDS padding: (pad_amount_code + 1) dwords inserted every
// (8 << pad_interval_code) bytes written.
__device__ __forceinline__ void tdm_load_2d_f16(const void* gptr, unsigned lds_off,
                                                unsigned dim0_elems, unsigned dim1_rows,
                                                unsigned stride0_elems,
                                                unsigned pad_interval_code,
                                                unsigned pad_amount_code,
                                                bool pad_en) {
#if HAVE_TDM
  unsigned long long ga = (unsigned long long)gptr;
  u32x4 g0;
  g0[0] = 1u;                                        // count=1, user descriptor
  g0[1] = lds_off;                                   // lds_addr (bytes)
  g0[2] = (unsigned)(ga & 0xffffffffu);              // global_addr[31:0]
  g0[3] = (unsigned)((ga >> 32) & 0x1ffffffu) | (2u << 30);  // addr[56:32], type=2
  i32x8 g1;
  unsigned w0 = (1u << 16);                          // data_size=1 -> 2 bytes
  if (pad_en) w0 |= (1u << 20) | (pad_interval_code << 22) | (pad_amount_code << 25);
  g1[0] = (int)w0;
  g1[1] = (int)((dim0_elems & 0xffffu) << 16);       // tensor_dim0[15:0] @ bits63:48
  g1[2] = (int)(((dim0_elems >> 16) & 0xffffu) | ((dim1_rows & 0xffffu) << 16));
  g1[3] = (int)(((dim1_rows >> 16) & 0xffffu) | ((dim0_elems & 0xffffu) << 16)); // tile_dim0
  g1[4] = (int)(dim1_rows & 0xffffu);                // tile_dim1; tile_dim2=0
  g1[5] = (int)stride0_elems;                        // tensor_dim0_stride[31:0]
  g1[6] = 0;                                         // stride0 hi / stride1 lo
  g1[7] = 0;
  i32x4 gz = {0, 0, 0, 0};
#if defined(__clang_major__) && (__clang_major__ >= 23)
  i32x8 gz8 = {0, 0, 0, 0, 0, 0, 0, 0};
  __builtin_amdgcn_tensor_load_to_lds(g0, g1, gz, gz, gz8, 0);
#else
  __builtin_amdgcn_tensor_load_to_lds(g0, g1, gz, gz, 0);
#endif
#else
  (void)gptr; (void)lds_off; (void)dim0_elems; (void)dim1_rows;
  (void)stride0_elems; (void)pad_interval_code; (void)pad_amount_code; (void)pad_en;
#endif
}

// ---------------- packing kernels ----------------

__global__ void pack_x_kernel(const float* __restrict__ x, _Float16* __restrict__ A1) {
  size_t i = (size_t)blockIdx.x * blockDim.x + threadIdx.x;
  if (i >= kSB * 128) return;
  int k = (int)(i & 127);
  size_t r = i >> 7;
  int b = (int)(r & (kB - 1));
  int s = (int)(r >> 7);
  float v = (k < kE) ? x[((size_t)b * kS + s) * kE + k] : 0.f;
  A1[i] = (_Float16)v;
}

__global__ void pack_pad_kernel(const float* __restrict__ src, _Float16* __restrict__ dst,
                                int R, int C, int Cp) {
  int i = blockIdx.x * blockDim.x + threadIdx.x;
  if (i >= R * Cp) return;
  int k = i % Cp;
  int r = i / Cp;
  dst[i] = (_Float16)((k < C) ? src[r * C + k] : 0.f);
}

__global__ void bias_sum_kernel(const float* __restrict__ b1, const float* __restrict__ b2,
                                float* __restrict__ dst, int n) {
  int i = blockIdx.x * blockDim.x + threadIdx.x;
  if (i < n) dst[i] = b1[i] + b2[i];
}

// ---------------- input-projection GEMM ----------------
// xg[dir][r][g] = A[r] @ W[dir][g]^T + bias.  grid: (SB/16, 1, 2), block: 1024.
// The 16-row A strip is TDM-staged into LDS once and shared by all 32 waves
// (each wave owns one 16-wide N-tile of the 512 gate outputs).
__global__ __launch_bounds__(1024) void xg_gemm_kernel(
    const _Float16* __restrict__ A,
    const _Float16* __restrict__ W,
    const float* __restrict__ bias,
    _Float16* __restrict__ xg, int Kpad) {
  __shared__ _Float16 atile[16 * 264];  // 16 rows x (Kpad+8) halves, max Kpad=256
  const int dir  = blockIdx.z;
  const int wave = threadIdx.x >> 5;
  const int n0   = wave * 16;
  const int m0   = blockIdx.x * 16;
  const int ldl  = Kpad + 8;
  const _Float16* Wd = W + (size_t)dir * kG * Kpad;

#if HAVE_TDM
  if (wave == 0) {
    tdm_load_2d_f16(A + (size_t)m0 * Kpad, (unsigned)(size_t)&atile[0],
                    (unsigned)Kpad, 16u, (unsigned)Kpad,
                    (Kpad == 128) ? 5u : 6u, 3u, true);   // pad 16B per row
    __builtin_amdgcn_s_wait_tensorcnt(0);
  }
#else
  for (int i = threadIdx.x; i < 16 * Kpad; i += 1024) {
    int r = i / Kpad, c = i % Kpad;
    atile[r * ldl + c] = A[(size_t)(m0 + r) * Kpad + c];
  }
#endif
  __syncthreads();

  v8f acc = zero8();
  for (int k0 = 0; k0 < Kpad; k0 += 32) {
    v16h a = load_a_frag(&atile[k0], ldl);            // LDS (ds_load_b128)
    v16h b = load_b_frag(Wd + (size_t)n0 * Kpad + k0, Kpad);
    acc = wmma_f16(a, b, acc);
  }
  const int lane = threadIdx.x & 31;
  const int N   = lane & 15;
  const int chi = (lane >> 4) << 3;
  const float bv = bias[dir * kG + n0 + N];
  _Float16* dst = xg + (size_t)dir * kSB * kG;
#pragma unroll
  for (int v = 0; v < 8; ++v)
    dst[(size_t)(m0 + v + chi) * kG + n0 + N] = (_Float16)(acc[v] + bv);
}

// ---------------- persistent LSTM recurrence: one block per direction ----------
// block = 1024 threads = 32 waves; each wave owns 2 h-tiles (16x16), c in regs,
// h kept in LDS (f16); Whh TDM-staged into LDS once (padded rows, stride 136)
// and reused for all 512 steps; xg precomputed.
__global__ __launch_bounds__(1024) void lstm_dir_kernel(
    const _Float16* __restrict__ xg,    // (2, S*B, 512)
    const _Float16* __restrict__ whh,   // (2, 512, 128)
    _Float16* __restrict__ outA) {      // (S*B, 256)
  __shared__ _Float16 hbuf[kB][kLPAD];   //  34,816 B
  __shared__ _Float16 wlds[kG][kLPAD];   // 139,264 B (<= 320KB/WGP per ISA)
  const int dir = blockIdx.x;
  const _Float16* xgd = xg + (size_t)dir * kSB * kG;
  const _Float16* wd  = whh + (size_t)dir * kG * kH;
  const int col_off = dir * kH;
  const int tid  = threadIdx.x;
  const int wave = tid >> 5;
  const int lane = tid & 31;
  const int N    = lane & 15;
  const int chi  = (lane >> 4) << 3;

  for (int i = tid; i < kB * kLPAD; i += 1024) (&hbuf[0][0])[i] = (_Float16)0.f;

  // stage Whh (512x128 f16) into LDS with 16B row padding -> stride kLPAD
#if HAVE_TDM
  if (wave == 0) {
    tdm_load_2d_f16(wd, (unsigned)(size_t)&wlds[0][0],
                    (unsigned)kH, (unsigned)kG, (unsigned)kH,
                    5u /*256B interval*/, 3u /*4 dwords*/, true);
    __builtin_amdgcn_s_wait_tensorcnt(0);
  }
#else
  for (int i = tid; i < kG * kH; i += 1024) {
    int r = i >> 7, c = i & (kH - 1);
    wlds[r][c] = wd[i];
  }
#endif

  float c_state[2][8];
#pragma unroll
  for (int t = 0; t < 2; ++t)
#pragma unroll
    for (int v = 0; v < 8; ++v) c_state[t][v] = 0.f;

  __syncthreads();

  for (int s = 0; s < kS; ++s) {
    const int se = dir ? (kS - 1 - s) : s;
    const _Float16* xrow = xgd + (size_t)se * kB * kG;
    if (s + 1 < kS) {  // prefetch next step's gate slab (global_prefetch_b8)
      const int sn = dir ? (kS - 2 - s) : (s + 1);
      __builtin_prefetch(xgd + (size_t)sn * kB * kG + (size_t)tid * 64, 0, 1);
    }
    float hval[2][8];
#pragma unroll
    for (int tt = 0; tt < 2; ++tt) {
      const int tile = wave + tt * 32;
      const int b0 = (tile >> 3) << 4;
      const int j0 = (tile & 7) << 4;
      v8f acc[4];
#pragma unroll
      for (int g = 0; g < 4; ++g) acc[g] = zero8();
#pragma unroll
      for (int k0 = 0; k0 < kH; k0 += 32) {
        v16h a = load_a_frag(&hbuf[b0][k0], kLPAD);          // LDS h(t-1)
#pragma unroll
        for (int g = 0; g < 4; ++g) {
          v16h b = load_b_frag(&wlds[g * kH + j0][k0], kLPAD);  // LDS Whh
          acc[g] = wmma_f16(a, b, acc[g]);
        }
      }
#pragma unroll
      for (int v = 0; v < 8; ++v) {
        const int bm = b0 + v + chi;
        const size_t xb = (size_t)bm * kG + j0 + N;
        const float gi = acc[0][v] + (float)xrow[xb + 0 * kH];
        const float gf = acc[1][v] + (float)xrow[xb + 1 * kH];
        const float gc = acc[2][v] + (float)xrow[xb + 2 * kH];
        const float go = acc[3][v] + (float)xrow[xb + 3 * kH];
        const float cc = sigmoidf_(gf) * c_state[tt][v] + sigmoidf_(gi) * tanhf(gc);
        c_state[tt][v] = cc;
        hval[tt][v] = sigmoidf_(go) * tanhf(cc);
      }
    }
    __syncthreads();  // all waves done reading h(t-1)
#pragma unroll
    for (int tt = 0; tt < 2; ++tt) {
      const int tile = wave + tt * 32;
      const int b0 = (tile >> 3) << 4;
      const int j0 = (tile & 7) << 4;
#pragma unroll
      for (int v = 0; v < 8; ++v) {
        const int bm = b0 + v + chi;
        const _Float16 h16 = (_Float16)hval[tt][v];
        hbuf[bm][j0 + N] = h16;
        outA[((size_t)se * kB + bm) * kH2 + col_off + j0 + N] = h16;
      }
    }
    __syncthreads();  // h(t) visible before next step
  }
}

// ---------------- FC: emis[b][s][t] = h2 @ fc_w^T + fc_b --------------------
__global__ void fc_kernel(const _Float16* __restrict__ A3,
                          const _Float16* __restrict__ fw,
                          const float* __restrict__ fb,
                          float* __restrict__ emis) {
  const int wave = threadIdx.x >> 5;
  const int m0 = blockIdx.x * 16;
  const int n0 = wave * 16;
  v8f acc = zero8();
#pragma unroll
  for (int k0 = 0; k0 < kH2; k0 += 32) {
    v16h a = load_a_frag(A3 + (size_t)m0 * kH2 + k0, kH2);
    v16h b = load_b_frag(fw + (size_t)n0 * kH2 + k0, kH2);
    acc = wmma_f16(a, b, acc);
  }
  const int lane = threadIdx.x & 31;
  const int N   = lane & 15;
  const int chi = (lane >> 4) << 3;
  const float bv = fb[n0 + N];
#pragma unroll
  for (int v = 0; v < 8; ++v) {
    const int r = m0 + v + chi;      // r = s*B + b
    const int srow = r >> 7;
    const int brow = r & (kB - 1);
    emis[((size_t)brow * kS + srow) * kT + n0 + N] = acc[v] + bv;
  }
}

// ---------------- CRF NLL: one wave per batch row, lane = tag ----------------
__global__ void crf_kernel(const float* __restrict__ emis,
                           const float* __restrict__ trans,
                           const float* __restrict__ startT,
                           const float* __restrict__ endT,
                           const int* __restrict__ target,
                           const int* __restrict__ slen,
                           float* __restrict__ part) {
  const int b = blockIdx.x;
  const int t = threadIdx.x;  // 0..31 == tag
  const float* eb = emis + (size_t)b * kS * kT;
  const int* tg = target + (size_t)b * kS;
  const int len = slen[b];

  float col[kT];
#pragma unroll
  for (int i = 0; i < kT; ++i) col[i] = trans[i * kT + t];

  float alpha = startT[t] + eb[t];
  for (int s = 1; s < len; ++s) {
    float av[kT];
    float m = -3.0e38f;
#pragma unroll
    for (int i = 0; i < kT; ++i) {
      float ai = __shfl(alpha, i, 32) + col[i];
      av[i] = ai;
      m = fmaxf(m, ai);
    }
    float ss = 0.f;
#pragma unroll
    for (int i = 0; i < kT; ++i) ss += __expf(av[i] - m);
    alpha = m + __logf(ss) + eb[(size_t)s * kT + t];
  }
  float v = alpha + endT[t];
  float m = v;
#pragma unroll
  for (int o = 16; o > 0; o >>= 1) m = fmaxf(m, __shfl_xor(m, o, 32));
  float e = __expf(v - m);
#pragma unroll
  for (int o = 16; o > 0; o >>= 1) e += __shfl_xor(e, o, 32);
  const float denom = m + __logf(e);
  float np = 0.f;
  for (int s = 1 + t; s < len; s += 32)
    np += eb[(size_t)s * kT + tg[s]] + trans[tg[s - 1] * kT + tg[s]];
#pragma unroll
  for (int o = 16; o > 0; o >>= 1) np += __shfl_xor(np, o, 32);
  if (t == 0) {
    float num = np + startT[tg[0]] + eb[tg[0]] + endT[tg[len - 1]];
    part[b] = num - denom;
  }
}

__global__ void final_reduce_kernel(const float* __restrict__ part, float* __restrict__ out) {
  __shared__ float sd[kB];
  const int t = threadIdx.x;
  sd[t] = part[t];
  __syncthreads();
  for (int o = 64; o > 0; o >>= 1) {
    if (t < o) sd[t] += sd[t + o];
    __syncthreads();
  }
  if (t == 0) out[0] = -sd[0];
}

// ---------------- host ----------------

extern "C" void kernel_launch(void* const* d_in, const int* in_sizes, int n_in,
                              void* d_out, int out_size, void* d_ws, size_t ws_size,
                              hipStream_t stream) {
  (void)in_sizes; (void)n_in; (void)out_size; (void)ws_size;
  const float* x        = (const float*)d_in[0];
  const float* l1f_wih  = (const float*)d_in[1];
  const float* l1f_whh  = (const float*)d_in[2];
  const float* l1f_bih  = (const float*)d_in[3];
  const float* l1f_bhh  = (const float*)d_in[4];
  const float* l1b_wih  = (const float*)d_in[5];
  const float* l1b_whh  = (const float*)d_in[6];
  const float* l1b_bih  = (const float*)d_in[7];
  const float* l1b_bhh  = (const float*)d_in[8];
  const float* l2f_wih  = (const float*)d_in[9];
  const float* l2f_whh  = (const float*)d_in[10];
  const float* l2f_bih  = (const float*)d_in[11];
  const float* l2f_bhh  = (const float*)d_in[12];
  const float* l2b_wih  = (const float*)d_in[13];
  const float* l2b_whh  = (const float*)d_in[14];
  const float* l2b_bih  = (const float*)d_in[15];
  const float* l2b_bhh  = (const float*)d_in[16];
  const float* fc_w     = (const float*)d_in[17];
  const float* fc_b     = (const float*)d_in[18];
  const float* trans    = (const float*)d_in[19];
  const float* startT   = (const float*)d_in[20];
  const float* endT     = (const float*)d_in[21];
  const int*   target   = (const int*)d_in[22];
  const int*   slen     = (const int*)d_in[23];

  char* base = (char*)d_ws;
  size_t off = 0;
  auto alloc = [&](size_t bytes) -> char* {
    off = (off + 255) & ~(size_t)255;
    char* p = base + off;
    off += bytes;
    return p;
  };

  _Float16* A1  = (_Float16*)alloc(kSB * 128 * 2);          // padded x, f16
  _Float16* A2  = (_Float16*)alloc(kSB * kH2 * 2);          // layer-1 output
  _Float16* A3  = (_Float16*)alloc(kSB * kH2 * 2);          // layer-2 output
  _Float16* XG  = (_Float16*)alloc(2 * kSB * kG * 2);       // gate pre-activations
  _Float16* W1  = (_Float16*)alloc((size_t)2 * kG * 128 * 2);
  _Float16* Wh1 = (_Float16*)alloc((size_t)2 * kG * kH * 2);
  _Float16* W2  = (_Float16*)alloc((size_t)2 * kG * kH2 * 2);
  _Float16* Wh2 = (_Float16*)alloc((size_t)2 * kG * kH * 2);
  float*    B1  = (float*)alloc(2 * kG * 4);
  float*    B2  = (float*)alloc(2 * kG * 4);
  _Float16* FW  = (_Float16*)alloc((size_t)kT * kH2 * 2);
  float*    EM  = (float*)alloc((size_t)kB * kS * kT * 4);
  float*    PT  = (float*)alloc(kB * 4);

  pack_x_kernel<<<(unsigned)(kSB * 128 / 256), 256, 0, stream>>>(x, A1);
  pack_pad_kernel<<<(kG * 128 + 255) / 256, 256, 0, stream>>>(l1f_wih, W1, kG, kE, 128);
  pack_pad_kernel<<<(kG * 128 + 255) / 256, 256, 0, stream>>>(l1b_wih, W1 + (size_t)kG * 128, kG, kE, 128);
  pack_pad_kernel<<<(kG * kH + 255) / 256, 256, 0, stream>>>(l1f_whh, Wh1, kG, kH, kH);
  pack_pad_kernel<<<(kG * kH + 255) / 256, 256, 0, stream>>>(l1b_whh, Wh1 + (size_t)kG * kH, kG, kH, kH);
  pack_pad_kernel<<<(kG * kH2 + 255) / 256, 256, 0, stream>>>(l2f_wih, W2, kG, kH2, kH2);
  pack_pad_kernel<<<(kG * kH2 + 255) / 256, 256, 0, stream>>>(l2b_wih, W2 + (size_t)kG * kH2, kG, kH2, kH2);
  pack_pad_kernel<<<(kG * kH + 255) / 256, 256, 0, stream>>>(l2f_whh, Wh2, kG, kH, kH);
  pack_pad_kernel<<<(kG * kH + 255) / 256, 256, 0, stream>>>(l2b_whh, Wh2 + (size_t)kG * kH, kG, kH, kH);
  pack_pad_kernel<<<(kT * kH2 + 255) / 256, 256, 0, stream>>>(fc_w, FW, kT, kH2, kH2);
  bias_sum_kernel<<<2, 256, 0, stream>>>(l1f_bih, l1f_bhh, B1, kG);
  bias_sum_kernel<<<2, 256, 0, stream>>>(l1b_bih, l1b_bhh, B1 + kG, kG);
  bias_sum_kernel<<<2, 256, 0, stream>>>(l2f_bih, l2f_bhh, B2, kG);
  bias_sum_kernel<<<2, 256, 0, stream>>>(l2b_bih, l2b_bhh, B2 + kG, kG);

  dim3 ggrid((unsigned)(kSB / 16), 1, 2);
  // layer 1
  xg_gemm_kernel<<<ggrid, 1024, 0, stream>>>(A1, W1, B1, XG, 128);
  lstm_dir_kernel<<<2, 1024, 0, stream>>>(XG, Wh1, A2);
  // layer 2
  xg_gemm_kernel<<<ggrid, 1024, 0, stream>>>(A2, W2, B2, XG, kH2);
  lstm_dir_kernel<<<2, 1024, 0, stream>>>(XG, Wh2, A3);
  // FC -> emissions
  fc_kernel<<<(unsigned)(kSB / 16), 64, 0, stream>>>(A3, FW, fc_b, EM);
  // CRF NLL
  crf_kernel<<<kB, 32, 0, stream>>>(EM, trans, startT, endT, target, slen, PT);
  final_reduce_kernel<<<1, kB, 0, stream>>>(PT, (float*)d_out);
}
